// SQEmbedding_3023656976729
// MI455X (gfx1250) — matrix operand order repaired and checked by reading, not compiled, not run on hardware
//
#include <hip/hip_runtime.h>
#include <stdint.h>

typedef unsigned short u16;
typedef __attribute__((ext_vector_type(16))) __bf16 v16bf;
typedef __attribute__((ext_vector_type(8)))  float  v8f;
typedef __attribute__((ext_vector_type(8)))  unsigned int v8u;

#define NROWS_TOT 32768     // B*S
#define MCODES    1024
#define DDIM      64
#define KAB       128       // concatenated K for distance GEMM
#define QOFF      2097152   // B*S*D

// LDS layout (dynamic):
#define OFF_DIST  0          // 16*1024 f32  = 65536 B (overlaid by qpart in phase C)
#define OFF_ENC   65536      // 16*1024 bf16 = 32768 B
#define OFF_AP    98304      // 16*128 bf16  = 4096 B
#define OFF_XX    102400     // 16 f32
#define OFF_SUME  102464     // 16 f32
#define OFF_WSUM  102528     // 8 f32
#define SMEM_BYTES 102560

__device__ __forceinline__ u16 f2bf(float f) {
    uint32_t u = __builtin_bit_cast(uint32_t, f);
    uint32_t r = u + 0x7FFFu + ((u >> 16) & 1u);   // round-to-nearest-even
    return (u16)(r >> 16);
}

__device__ __forceinline__ v16bf pack16(uint4 lo, uint4 hi) {
    v8u t;
    t[0] = lo.x; t[1] = lo.y; t[2] = lo.z; t[3] = lo.w;
    t[4] = hi.x; t[5] = hi.y; t[6] = hi.z; t[7] = hi.w;
    return __builtin_bit_cast(v16bf, t);
}

__device__ __forceinline__ v8f wmma_bf16(v16bf a, v16bf b, v8f c) {
    return __builtin_amdgcn_wmma_f32_16x16x32_bf16(
        false, a, false, b, (short)0, c, false, false);
}

// ---------------------------------------------------------------------------
// Prep: Bmat[m][k] = (k<64 ? -E[m][k] : 0.5*E[m][k-64]^2)  (1024x128 bf16)
//       Ebt[d][k]  = E[k][d]                                (64x1024 bf16)
//       zero counts + loss accumulator
// ---------------------------------------------------------------------------
extern "C" __global__ __launch_bounds__(256)
void sq_prep(const float* __restrict__ E, u16* __restrict__ Bmat,
             u16* __restrict__ Ebt, int* __restrict__ counts,
             float* __restrict__ lossAcc) {
    int i = blockIdx.x * 256 + threadIdx.x;
    if (i < MCODES * KAB) {
        int m = i >> 7, k = i & 127;
        float e = E[m * DDIM + (k & 63)];
        Bmat[i] = f2bf((k < 64) ? -e : 0.5f * e * e);
    }
    if (i < DDIM * MCODES) {
        int d = i >> 10, k = i & 1023;
        Ebt[i] = f2bf(E[k * DDIM + d]);
    }
    if (i < MCODES) counts[i] = 0;
    if (i == 0) *lossAcc = 0.0f;
}

// ---------------------------------------------------------------------------
// Main fused kernel: one workgroup per 16 rows.
// ---------------------------------------------------------------------------
extern "C" __global__ __launch_bounds__(256)
void sq_main(const float* __restrict__ X, const float* __restrict__ LV,
             const float* __restrict__ noise, const float* __restrict__ Tptr,
             const u16* __restrict__ Bmat, const u16* __restrict__ Ebt,
             float* __restrict__ out, int* __restrict__ counts,
             float* __restrict__ lossAcc) {
    extern __shared__ char smem[];
    float* dist = (float*)(smem + OFF_DIST);
    u16*   enc  = (u16*)  (smem + OFF_ENC);
    u16*   Ap   = (u16*)  (smem + OFF_AP);
    float* xx   = (float*)(smem + OFF_XX);
    float* sumE = (float*)(smem + OFF_SUME);
    float* wsum = (float*)(smem + OFF_WSUM);

    const int tid  = threadIdx.x;
    const int lane = tid & 31;
    const int wave = tid >> 5;
    const int n0   = blockIdx.x * 16;
    const float invT = 1.0f / Tptr[0];
    const float EPS = 1.1920929e-7f;

    // Warm the block's 64 KB gumbel-noise tile (dominant HBM stream) into
    // cache while the WMMA GEMM runs: 2 x 128B cachelines per thread.
    {
        const char* nbase = (const char*)(noise + (size_t)n0 * MCODES);
        __builtin_prefetch(nbase + (size_t)tid * 256, 0, 3);
        __builtin_prefetch(nbase + (size_t)tid * 256 + 128, 0, 3);
    }

    // ---------------- Phase A0: A-panel = [p*x | p], xx = 0.5*sum(p*x*x) ----
    {
        int row = tid >> 4;        // 16 threads per row
        int l16 = tid & 15;
        const float* xr = X  + (size_t)(n0 + row) * DDIM;
        const float* lr = LV + (size_t)(n0 + row) * DDIM;
        float xxp = 0.0f;
        #pragma unroll
        for (int j = 0; j < 4; ++j) {
            int d = l16 + 16 * j;
            float xv = xr[d];
            float p  = __expf(-lr[d]);
            float px = p * xv;
            Ap[row * KAB + d]      = f2bf(px);
            Ap[row * KAB + 64 + d] = f2bf(p);
            xxp += 0.5f * px * xv;
        }
        #pragma unroll
        for (int off = 8; off > 0; off >>= 1) xxp += __shfl_xor(xxp, off, 32);
        if (l16 == 0) xx[row] = xxp;
    }
    __syncthreads();

    // ---------------- Phase A: distances[16][1024] via bf16 WMMA -----------
    {
        const int r   = lane & 15;
        const int hi  = lane >> 4;          // 0 or 1
        const int ka  = hi * 8;             // unused (A also uses contiguous kb16)
        const int kb16 = hi * 16;           // B contiguous offset
        v16bf a[4];
        #pragma unroll
        for (int c = 0; c < 4; ++c) {
            const uint4* p0 = (const uint4*)&Ap[r * KAB + c * 32 + kb16];
            const uint4* p1 = (const uint4*)&Ap[r * KAB + c * 32 + kb16 + 8];
            a[c] = pack16(*p0, *p1);
        }
        float xrow[8];
        #pragma unroll
        for (int j = 0; j < 8; ++j) xrow[j] = xx[j + hi * 8];

        for (int i = 0; i < 8; ++i) {
            int m0 = (wave * 8 + i) * 16;
            v8f acc = {};
            #pragma unroll
            for (int c = 0; c < 4; ++c) {
                const uint4* q0 = (const uint4*)&Bmat[(size_t)(m0 + r) * KAB + c * 32 + kb16];
                v16bf b = pack16(q0[0], q0[1]);
                acc = wmma_bf16(a[c], b, acc);
            }
            int col = m0 + r;
            #pragma unroll
            for (int j = 0; j < 8; ++j)
                dist[(j + hi * 8) * MCODES + col] = acc[j] + xrow[j];
        }
    }
    __syncthreads();

    // ---------------- Phase B: argmin, gumbel softmax (unnorm), entropy ----
    for (int rr = 0; rr < 2; ++rr) {
        int r = wave * 2 + rr;
        const float* nrow = noise + (size_t)(n0 + r) * MCODES;
        float minD = 3.4e38f, smax = -3.4e38f;
        int argm = 0;
        for (int i = 0; i < 32; ++i) {
            int m = i * 32 + lane;
            float d = dist[r * MCODES + m];
            if (d < minD) { minD = d; argm = m; }
            float u = fminf(fmaxf(nrow[m], EPS), 1.0f - EPS);
            float g = -__logf(-__logf(u));
            smax = fmaxf(smax, (g - d) * invT);
        }
        #pragma unroll
        for (int off = 16; off > 0; off >>= 1) {
            float oD = __shfl_xor(minD, off, 32);
            int   oA = __shfl_xor(argm, off, 32);
            if (oD < minD || (oD == minD && oA < argm)) { minD = oD; argm = oA; }
            smax = fmaxf(smax, __shfl_xor(smax, off, 32));
        }
        float max0 = -minD;
        float Z0 = 0.0f, S1 = 0.0f, sE = 0.0f;
        for (int i = 0; i < 32; ++i) {
            int m = i * 32 + lane;
            float d = dist[r * MCODES + m];
            float t0 = -d - max0;
            float e0 = __expf(t0);
            Z0 += e0; S1 += e0 * t0;
            float u = fminf(fmaxf(nrow[m], EPS), 1.0f - EPS);
            float g = -__logf(-__logf(u));
            float e1 = __expf((g - d) * invT - smax);
            sE += e1;
            enc[r * MCODES + m] = f2bf(e1);
        }
        #pragma unroll
        for (int off = 16; off > 0; off >>= 1) {
            Z0 += __shfl_xor(Z0, off, 32);
            S1 += __shfl_xor(S1, off, 32);
            sE += __shfl_xor(sE, off, 32);
        }
        if (lane == 0) {
            sumE[r] = sE;
            atomicAdd(&counts[argm], 1);
            atomicAdd(lossAcc, S1 / Z0 - __logf(Z0));   // neg-entropy term
        }
    }
    __syncthreads();

    // ---------------- Phase C: quantized = (enc @ E) / sumE  (K-split) -----
    {
        float* qpart = dist;                // overlay dead distance buffer
        const int r    = lane & 15;
        const int hi   = lane >> 4;
        const int ka   = hi * 8;
        const int kb16 = hi * 16;
        #pragma unroll
        for (int t = 0; t < 4; ++t) {       // 4 col-tiles of 16 (D = 64)
            v8f acc = {};
            #pragma unroll
            for (int c = 0; c < 4; ++c) {   // this wave's K slice: 128
                int kb = wave * 128 + c * 32;
                const uint4* a0 = (const uint4*)&enc[r * MCODES + kb + kb16];
                const uint4* a1 = (const uint4*)&enc[r * MCODES + kb + kb16 + 8];
                const uint4* b0 = (const uint4*)&Ebt[(size_t)(t * 16 + r) * MCODES + kb + kb16];
                acc = wmma_bf16(pack16(*a0, *a1), pack16(b0[0], b0[1]), acc);
            }
            #pragma unroll
            for (int j = 0; j < 8; ++j)
                qpart[wave * 1024 + (j + hi * 8) * DDIM + t * 16 + r] = acc[j];
        }
    }
    __syncthreads();

    // ---------------- Reduce partials, write quantized, recon loss ---------
    {
        const float* qpart = dist;
        float racc = 0.0f;
        for (int idx = tid; idx < 16 * DDIM; idx += 256) {
            int row = idx >> 6, col = idx & 63;
            float q = 0.0f;
            #pragma unroll
            for (int w = 0; w < 8; ++w) q += qpart[w * 1024 + row * DDIM + col];
            q /= sumE[row];
            size_t gi = (size_t)(n0 + row) * DDIM + col;
            out[gi] = q;
            float dq = X[gi] - q;
            racc += 0.5f * __expf(-LV[gi]) * dq * dq;
        }
        #pragma unroll
        for (int off = 16; off > 0; off >>= 1) racc += __shfl_xor(racc, off, 32);
        if (lane == 0) wsum[wave] = racc;
        __syncthreads();
        if (tid == 0) {
            float s = 0.0f;
            #pragma unroll
            for (int w = 0; w < 8; ++w) s += wsum[w];
            atomicAdd(lossAcc, s);
        }
    }
}

// ---------------------------------------------------------------------------
// Final: loss = acc / B ; perplexity from argmin histogram.
// ---------------------------------------------------------------------------
extern "C" __global__ __launch_bounds__(1024)
void sq_final(const int* __restrict__ counts, const float* __restrict__ lossAcc,
              float* __restrict__ out2) {
    __shared__ float red[32];
    int tid = threadIdx.x;
    float avg = (float)counts[tid] * (1.0f / 32768.0f);
    float term = avg * __logf(avg + 1e-10f);
    #pragma unroll
    for (int off = 16; off > 0; off >>= 1) term += __shfl_xor(term, off, 32);
    if ((tid & 31) == 0) red[tid >> 5] = term;
    __syncthreads();
    if (tid < 32) {
        float s = red[tid];
        #pragma unroll
        for (int off = 16; off > 0; off >>= 1) s += __shfl_xor(s, off, 32);
        if (tid == 0) {
            out2[0] = lossAcc[0] * (1.0f / 32.0f);
            out2[1] = __expf(-s);
        }
    }
}

extern "C" void kernel_launch(void* const* d_in, const int* in_sizes, int n_in,
                              void* d_out, int out_size, void* d_ws, size_t ws_size,
                              hipStream_t stream) {
    const float* X     = (const float*)d_in[0];
    const float* LV    = (const float*)d_in[1];
    const float* E     = (const float*)d_in[2];
    const float* noise = (const float*)d_in[3];
    const float* T     = (const float*)d_in[4];
    float* out = (float*)d_out;

    uint8_t* ws = (uint8_t*)d_ws;
    u16*   Bmat    = (u16*)ws;                              // 1024*128*2 = 262144 B
    u16*   Ebt     = (u16*)(ws + 262144);                   // 64*1024*2  = 131072 B
    int*   counts  = (int*)(ws + 262144 + 131072);          // 4096 B
    float* lossAcc = (float*)(ws + 262144 + 131072 + 4096); // 4 B

    sq_prep<<<512, 256, 0, stream>>>(E, Bmat, Ebt, counts, lossAcc);
    sq_main<<<NROWS_TOT / 16, 256, SMEM_BYTES, stream>>>(
        X, LV, noise, T, Bmat, Ebt, out, counts, lossAcc);
    sq_final<<<1, 1024, 0, stream>>>(counts, lossAcc, out + QOFF);
}
